// ControllerNetwork_32289564131801
// MI455X (gfx1250) — matrix operand order repaired and checked by reading, not compiled, not run on hardware
//
#include <hip/hip_runtime.h>
#include <hip/hip_bf16.h>
#include <math.h>

#define HIDDEN     2048
#define GATES      8192   // 4*HIDDEN
#define IN_DIM     80     // NUM_LAYERS + NUM_TYPES
#define NUM_TYPES  16
#define NUM_LAYERS 64
#define N_STEPS    128
#define TANH_CONST 2.5f
#define INV_TEMP   0.2f   // 1/5.0

typedef __attribute__((ext_vector_type(16))) __bf16 v16bf;
typedef __attribute__((ext_vector_type(8)))  float  v8f;

union BF16x16 {
    v16bf v;
    __bf16 e[16];
    uint4 q[2];
};

// ---------------------------------------------------------------------------
// workspace layout:
//   floats [0,2048)       h
//   floats [2048,4096)    c
//   floats [4096,6144)    out = relu(h)
//   floats [6144,14336)   gates (8192)
//   floats [14336,14464)  x (80 used, padded)
//   floats [14464,14544)  z (80 head pre-activations)
//   byte offset 65536:    swizzled bf16 W_hh (32 MB), if ws_size permits
// ---------------------------------------------------------------------------
#define WS_STATE_FLOATS 14544
#define WS_BW_OFFSET    65536

__global__ void init_ws_kernel(float* ws) {
    int i = blockIdx.x * blockDim.x + threadIdx.x;
    if (i < WS_STATE_FLOATS) ws[i] = 0.0f;
}

// One-time: swizzle W_hh (f32 row-major) into per-lane WMMA A-tile bf16 layout.
// Dest element index = ((tile*64 + chunk)*32 + lane)*16 + e, where
//   row  = tile*16 + (lane&15), g = (lane>>4)*8,
//   K    = chunk*32 + ((e<8) ? g+e : g+8+e).
// Each thread produces one 32-bit word (2 adjacent bf16).
__global__ void swizzle_whh_kernel(const float* __restrict__ W,
                                   unsigned* __restrict__ bw_words) {
    unsigned w = blockIdx.x * blockDim.x + threadIdx.x;  // < 8,388,608
    if (w >= (GATES / 16) * 16384u / 2u * 2u) { }        // 512 tiles * 16384 words/... 
    if (w >= 8388608u) return;
    unsigned wi    = w & 7u;
    unsigned lane  = (w >> 3) & 31u;
    unsigned chunk = (w >> 8) & 63u;
    unsigned tile  = w >> 14;
    unsigned e0    = wi * 2u;
    unsigned g     = (lane >> 4) * 8u;
    unsigned koff  = (e0 < 8u) ? (g + e0) : (g + 8u + e0);
    unsigned row   = tile * 16u + (lane & 15u);
    const float* s = W + (size_t)row * HIDDEN + chunk * 32u + koff;
    union { __bf16 e[2]; unsigned u; } p;
    p.e[0] = (__bf16)s[0];
    p.e[1] = (__bf16)s[1];
    bw_words[w] = p.u;
}

// ---------------------------------------------------------------------------
// Shared epilogue for the gates kernels: W_ih @ x + biases + store
// ---------------------------------------------------------------------------
__device__ __forceinline__ void gates_epilogue(
        v8f acc, int lane, int half, int m0, int mrow,
        const float* __restrict__ W_ih, const float* __restrict__ b_ih,
        const float* __restrict__ b_hh, const float* __restrict__ x,
        float* __restrict__ gates) {
    float p = 0.0f;
    {
        const float4* wr = (const float4*)(W_ih + (size_t)mrow * IN_DIM + half * 40);
        const float4* xr = (const float4*)(x + half * 40);
        #pragma unroll
        for (int j = 0; j < 10; ++j) {
            float4 wv = wr[j], xv = xr[j];
            p = fmaf(wv.x, xv.x, p);
            p = fmaf(wv.y, xv.y, p);
            p = fmaf(wv.z, xv.z, p);
            p = fmaf(wv.w, xv.w, p);
        }
    }
    p += __shfl_xor(p, 16, 32);              // full 80-dot in both halves
    p += b_ih[mrow] + b_hh[mrow];            // row = m0 + (lane&15)

    #pragma unroll
    for (int r = 0; r < 8; ++r) {
        float dlo = __shfl(p, r,     32);    // extra term for row m0+r
        float dhi = __shfl(p, 8 + r, 32);    // extra term for row m0+8+r
        if (lane == 0)  gates[m0 + r]     = acc[r] + dlo;
        if (lane == 16) gates[m0 + 8 + r] = acc[r] + dhi;
    }
}

// ---------------------------------------------------------------------------
// Fast path: pre-swizzled bf16 weights. 128 blocks x 128 threads, wave = 16 rows.
// ---------------------------------------------------------------------------
__global__ void __launch_bounds__(128)
gates_kernel_pre(const __bf16* __restrict__ bw, const float* __restrict__ W_ih,
                 const float* __restrict__ b_ih, const float* __restrict__ b_hh,
                 const float* __restrict__ h,    const float* __restrict__ x,
                 float* __restrict__ gates) {
    __shared__ __align__(16) __bf16 lh[HIDDEN];
    for (int i = threadIdx.x; i < HIDDEN; i += blockDim.x)
        lh[i] = (__bf16)h[i];
    __syncthreads();

    const int lane   = threadIdx.x & 31;
    const int waveId = (blockIdx.x * blockDim.x + threadIdx.x) >> 5;
    const int m0     = waveId * 16;
    const int half   = lane >> 4;
    const int mrow   = m0 + (lane & 15);

    const __bf16* ap = bw + (((size_t)waveId * 64) * 32 + lane) * 16;

    v8f acc = {};
    #pragma unroll 4
    for (int chunk = 0; chunk < 64; ++chunk) {
        BF16x16 A, B;
        A.q[0] = *(const uint4*)(ap);
        A.q[1] = *(const uint4*)(ap + 8);
        const int base = chunk * 32 + half * 8;
        B.q[0] = *(const uint4*)(lh + base);
        B.q[1] = *(const uint4*)(lh + base + 16);
        acc = __builtin_amdgcn_wmma_f32_16x16x32_bf16(
                  false, A.v, false, B.v, (short)0, acc, false, false);
        ap += 32 * 16;
    }

    gates_epilogue(acc, lane, half, m0, mrow, W_ih, b_ih, b_hh, x, gates);
}

// ---------------------------------------------------------------------------
// Fallback: read f32 weights, convert in-register (used if ws too small).
// ---------------------------------------------------------------------------
__global__ void __launch_bounds__(128)
gates_kernel_cvt(const float* __restrict__ W_hh, const float* __restrict__ W_ih,
                 const float* __restrict__ b_ih, const float* __restrict__ b_hh,
                 const float* __restrict__ h,    const float* __restrict__ x,
                 float* __restrict__ gates) {
    __shared__ __align__(16) __bf16 lh[HIDDEN];
    for (int i = threadIdx.x; i < HIDDEN; i += blockDim.x)
        lh[i] = (__bf16)h[i];
    __syncthreads();

    const int lane   = threadIdx.x & 31;
    const int waveId = (blockIdx.x * blockDim.x + threadIdx.x) >> 5;
    const int m0     = waveId * 16;
    const int half   = lane >> 4;
    const int mrow   = m0 + (lane & 15);
    const int g      = half * 8;

    const float* __restrict__ wbase = W_hh + (size_t)mrow * HIDDEN;

    v8f acc = {};
    for (int k0 = 0; k0 < HIDDEN; k0 += 32) {
        const float* wp = wbase + k0 + g;
        float4 w0 = *(const float4*)(wp +  0);
        float4 w1 = *(const float4*)(wp +  4);
        float4 w2 = *(const float4*)(wp + 16);
        float4 w3 = *(const float4*)(wp + 20);
        BF16x16 A;
        A.e[0]  = (__bf16)w0.x; A.e[1]  = (__bf16)w0.y;
        A.e[2]  = (__bf16)w0.z; A.e[3]  = (__bf16)w0.w;
        A.e[4]  = (__bf16)w1.x; A.e[5]  = (__bf16)w1.y;
        A.e[6]  = (__bf16)w1.z; A.e[7]  = (__bf16)w1.w;
        A.e[8]  = (__bf16)w2.x; A.e[9]  = (__bf16)w2.y;
        A.e[10] = (__bf16)w2.z; A.e[11] = (__bf16)w2.w;
        A.e[12] = (__bf16)w3.x; A.e[13] = (__bf16)w3.y;
        A.e[14] = (__bf16)w3.z; A.e[15] = (__bf16)w3.w;

        const int base = k0 + half * 8;
        BF16x16 B;
        B.q[0] = *(const uint4*)(lh + base);
        B.q[1] = *(const uint4*)(lh + base + 16);

        acc = __builtin_amdgcn_wmma_f32_16x16x32_bf16(
                  false, A.v, false, B.v, (short)0, acc, false, false);
    }

    gates_epilogue(acc, lane, half, m0, mrow, W_ih, b_ih, b_hh, x, gates);
}

__device__ __forceinline__ float sigmoidf(float v) {
    return 1.0f / (1.0f + expf(-v));
}

// LSTM cell elementwise update: grid 8 x 256 = 2048
__global__ void cell_kernel(const float* __restrict__ gates,
                            float* __restrict__ c, float* __restrict__ h,
                            float* __restrict__ outv) {
    int i = blockIdx.x * blockDim.x + threadIdx.x;
    if (i >= HIDDEN) return;
    float gi = sigmoidf(gates[i]);
    float gf = sigmoidf(gates[HIDDEN + i]);
    float gg = tanhf(gates[2 * HIDDEN + i]);
    float go = sigmoidf(gates[3 * HIDDEN + i]);
    float cn = gf * c[i] + gi * gg;
    float hn = go * tanhf(cn);
    c[i] = cn;
    h[i] = hn;
    outv[i] = fmaxf(hn, 0.0f);
}

// heads dot-products: 20 blocks x 128 threads = 80 waves; wave w -> row w.
__global__ void __launch_bounds__(128)
head_dots_kernel(const float* __restrict__ fc1_w, const float* __restrict__ fc1_b,
                 const float* __restrict__ fc2_w, const float* __restrict__ fc2_b,
                 const float* __restrict__ outv, float* __restrict__ z) {
    const int lane = threadIdx.x & 31;
    const int r    = (blockIdx.x * blockDim.x + threadIdx.x) >> 5;
    if (r >= IN_DIM) return;
    const float* wr = (r < NUM_TYPES)
                        ? (fc1_w + (size_t)r * HIDDEN)
                        : (fc2_w + (size_t)(r - NUM_TYPES) * HIDDEN);
    float bias = (r < NUM_TYPES) ? fc1_b[r] : fc2_b[r - NUM_TYPES];
    const float4* w4 = (const float4*)wr;
    const float4* o4 = (const float4*)outv;
    float p = 0.0f;
    #pragma unroll 4
    for (int e = lane; e < HIDDEN / 4; e += 32) {
        float4 wv = w4[e], ov = o4[e];
        p = fmaf(wv.x, ov.x, p);
        p = fmaf(wv.y, ov.y, p);
        p = fmaf(wv.z, ov.z, p);
        p = fmaf(wv.w, ov.w, p);
    }
    #pragma unroll
    for (int off = 16; off > 0; off >>= 1) p += __shfl_xor(p, off, 32);
    if (lane == 0) z[r] = p + bias;
}

__device__ __forceinline__ float rng_uniform(unsigned a, unsigned b) {
    unsigned hsh = a * 0x9E3779B1u + b * 0x85EBCA77u + 0x27220A95u;
    hsh ^= hsh >> 15; hsh *= 0x2C1B3C6Du;
    hsh ^= hsh >> 12; hsh *= 0x297A2D39u;
    hsh ^= hsh >> 15;
    float u = (float)(hsh >> 8) * (1.0f / 16777216.0f);
    return fminf(fmaxf(u, 1e-7f), 1.0f - 1e-7f);
}

// sampling: 1 block x 128 threads
__global__ void __launch_bounds__(128)
sample_kernel(const float* __restrict__ z, float* __restrict__ x,
              float* __restrict__ dout, int step) {
    const int tid = threadIdx.x;

    // ---- head 2: connection sampler (64 lanes)
    if (tid < NUM_LAYERS) {
        float zz = z[NUM_TYPES + tid];
        float cconst = (float)(step / 2);
        float soft2 = -log1pf(cconst * expf(-zz));     // log-prob
        float prob  = expf(soft2);
        float u = rng_uniform((unsigned)(step * 2 + 1), (unsigned)(1000 + tid));
        float t2 = ((u < prob) && (tid < step / 2)) ? 1.0f : 0.0f;
        dout[2048 + step * NUM_LAYERS + tid]  = soft2;  // s2 section
        dout[12288 + step * NUM_LAYERS + tid] = t2;     // t2 section
        x[NUM_TYPES + tid] = t2;
    }

    // ---- head 1: tempered-tanh softmax + Gumbel-max categorical (tiny)
    if (tid == 64) {
        float l[NUM_TYPES];
        float m = -1e30f;
        #pragma unroll
        for (int i = 0; i < NUM_TYPES; ++i) {
            l[i] = TANH_CONST * tanhf(z[i] * INV_TEMP);
            m = fmaxf(m, l[i]);
        }
        float se = 0.0f;
        #pragma unroll
        for (int i = 0; i < NUM_TYPES; ++i) se += expf(l[i] - m);
        float lse = logf(se) + m;
        int idx = 0;
        float best = -1e30f;
        #pragma unroll
        for (int i = 0; i < NUM_TYPES; ++i) {
            dout[step * NUM_TYPES + i] = l[i] - lse;    // s1 section
            float u = rng_uniform((unsigned)(step * 2), (unsigned)i);
            float score = l[i] - logf(-logf(u));        // Gumbel-max
            if (score > best) { best = score; idx = i; }
        }
        #pragma unroll
        for (int i = 0; i < NUM_TYPES; ++i) {
            float t1 = (i == idx) ? 1.0f : 0.0f;
            dout[10240 + step * NUM_TYPES + i] = t1;    // t1 section
            x[i] = t1;
        }
    }
}

extern "C" void kernel_launch(void* const* d_in, const int* in_sizes, int n_in,
                              void* d_out, int out_size, void* d_ws, size_t ws_size,
                              hipStream_t stream) {
    const float* W_ih  = (const float*)d_in[0];
    const float* W_hh  = (const float*)d_in[1];
    const float* b_ih  = (const float*)d_in[2];
    const float* b_hh  = (const float*)d_in[3];
    const float* fc1_w = (const float*)d_in[4];
    const float* fc1_b = (const float*)d_in[5];
    const float* fc2_w = (const float*)d_in[6];
    const float* fc2_b = (const float*)d_in[7];

    float* ws    = (float*)d_ws;
    float* h     = ws;
    float* c     = ws + 2048;
    float* outv  = ws + 4096;
    float* gates = ws + 6144;
    float* x     = ws + 14336;
    float* z     = ws + 14464;
    float* out   = (float*)d_out;

    const size_t bw_bytes = (size_t)GATES * HIDDEN * 2;           // 32 MB
    const bool   use_pre  = ws_size >= (WS_BW_OFFSET + bw_bytes);
    __bf16* bw = (__bf16*)((char*)d_ws + WS_BW_OFFSET);

    init_ws_kernel<<<57, 256, 0, stream>>>(ws);
    if (use_pre) {
        // 8,388,608 dest words / 256 = 32768 blocks
        swizzle_whh_kernel<<<32768, 256, 0, stream>>>(W_hh, (unsigned*)bw);
    }

    for (int step = 0; step < N_STEPS; ++step) {
        if (use_pre) {
            gates_kernel_pre<<<128, 128, 0, stream>>>(bw, W_ih, b_ih, b_hh,
                                                      h, x, gates);
        } else {
            gates_kernel_cvt<<<128, 128, 0, stream>>>(W_hh, W_ih, b_ih, b_hh,
                                                      h, x, gates);
        }
        cell_kernel<<<8, 256, 0, stream>>>(gates, c, h, outv);
        head_dots_kernel<<<20, 128, 0, stream>>>(fc1_w, fc1_b, fc2_w, fc2_b,
                                                 outv, z);
        sample_kernel<<<1, 128, 0, stream>>>(z, x, out, step);
    }
}
